// TitansMemory_2473901163033
// MI455X (gfx1250) — matrix operand order repaired and checked by reading, not compiled, not run on hardware
//
#include <hip/hip_runtime.h>
#include <hip/hip_bf16.h>
#include <math.h>
#include <stdint.h>

// ---------------------------------------------------------------------------
// Titans memory update for MI455X (gfx1250).
// - every GEMM via v_wmma_f32_16x16x32_bf16 (f32 accumulate)
// - double-buffered LDS staging, BK=64, one barrier per K-step
// - async global->LDS copies (ASYNCcnt) where layout allows
// - split-K (8) for the K=8192 weight-grad GEMMs + fused state update
// ---------------------------------------------------------------------------

typedef __bf16 bf16_t;
typedef bf16_t v16bf __attribute__((ext_vector_type(16)));
typedef bf16_t v8bf  __attribute__((ext_vector_type(8)));
typedef float  v8f   __attribute__((ext_vector_type(8)));
typedef int    v4i_vs __attribute__((vector_size(16)));   // builtin's pointee type

constexpr int BB = 2, TT = 4096, DD = 1024, MM = 512, KK = 3;
constexpr int NN = BB * TT;       // 8192 tokens
constexpr int SPLITK = 8;         // split-K for grad GEMMs
constexpr int RCH = 64;           // row chunks for column-sum

#if __has_builtin(__builtin_amdgcn_global_load_async_to_lds_b128) && \
    __has_builtin(__builtin_amdgcn_s_wait_asynccnt)
#define HAVE_ASYNC 1
__device__ __forceinline__ void async_cp16(const bf16_t* g, bf16_t* l) {
  // ROCm clang-22 declares the async-LDS builtins with generic int4* pointers.
  __builtin_amdgcn_global_load_async_to_lds_b128(
      (v4i_vs*)const_cast<bf16_t*>(g), (v4i_vs*)l, 0, 0);
}
#else
#define HAVE_ASYNC 0
#endif

// ------------------------------ prep kernels -------------------------------

// column partial sums over 64 row chunks (deterministic; no float atomics)
__global__ void colsum_kernel(const float* __restrict__ x, float* __restrict__ part) {
  int c = blockIdx.x * blockDim.x + threadIdx.x;   // column
  int rc = blockIdx.y;                             // row chunk
  const int rows = NN / RCH;
  float s = 0.f;
  for (int r = rc * rows; r < (rc + 1) * rows; ++r) s += x[(size_t)r * DD + c];
  part[(size_t)rc * DD + c] = s;
}

__global__ void scalars_kernel(const float* __restrict__ xpart,
                               const float* paw, const float* pab,
                               const float* pew, const float* peb,
                               const float* ptw, const float* ptb,
                               float* __restrict__ scal) {
  __shared__ float ra[256], re[256], rt[256];
  int t = threadIdx.x;
  float sa = 0.f, se = 0.f, st = 0.f;
  for (int i = t; i < DD; i += 256) {
    float xs = 0.f;
    for (int r = 0; r < RCH; ++r) xs += xpart[(size_t)r * DD + i];
    float xm = xs * (1.f / (float)NN);
    sa += paw[i] * xm; se += pew[i] * xm; st += ptw[i] * xm;
  }
  ra[t] = sa; re[t] = se; rt[t] = st;
  __syncthreads();
  for (int off = 128; off > 0; off >>= 1) {
    if (t < off) { ra[t] += ra[t+off]; re[t] += re[t+off]; rt[t] += rt[t+off]; }
    __syncthreads();
  }
  if (t == 0) {
    float a  = 1.f / (1.f + expf(-(ra[0] + pab[0])));   // sigmoid -> alpha
    float e  = 1.f / (1.f + expf(-(re[0] + peb[0])));   // sigmoid -> eta
    float th = log1pf(expf(rt[0] + ptb[0]));            // softplus -> theta
    scal[0] = a; scal[1] = e; scal[2] = th; scal[3] = 1.f - a;
  }
}

// depthwise conv over time (K=3, pad=1, per-batch boundaries), emit bf16
__global__ void conv_bf16_kernel(const float* __restrict__ x,
                                 const float* __restrict__ w,
                                 bf16_t* __restrict__ out) {
  size_t idx = (size_t)blockIdx.x * blockDim.x + threadIdx.x;
  if (idx >= (size_t)NN * DD) return;
  int d = (int)(idx % DD);
  int n = (int)(idx / DD);
  int t = n % TT;
  const float* xr = x + idx;
  float acc = w[d * KK + 1] * xr[0];
  if (t > 0)      acc += w[d * KK + 0] * xr[-(ptrdiff_t)DD];
  if (t < TT - 1) acc += w[d * KK + 2] * xr[DD];
  out[idx] = (bf16_t)acc;
}

__global__ void f32_to_bf16_kernel(const float* __restrict__ s,
                                   bf16_t* __restrict__ d, int n) {
  int i = blockIdx.x * blockDim.x + threadIdx.x;
  if (i < n) d[i] = (bf16_t)s[i];
}

// sum split-K partials, apply S_t = eta*S - theta*g ; W_t = (1-alpha)*W + S_t
__global__ void update_reduce_kernel(const float* __restrict__ part, size_t elems,
                                     const float* __restrict__ w_old,
                                     const float* __restrict__ m_old,
                                     float* __restrict__ nw, float* __restrict__ nm,
                                     const float* __restrict__ scal) {
  size_t i = (size_t)blockIdx.x * blockDim.x + threadIdx.x;
  if (i >= elems) return;
  float g = 0.f;
#pragma unroll
  for (int p = 0; p < SPLITK; ++p) g += part[(size_t)p * elems + i];
  float eta = scal[1], theta = scal[2], decay = scal[3];
  float nmv = eta * m_old[i] - theta * g;
  nm[i] = nmv;
  nw[i] = decay * w_old[i] + nmv;
}

// ------------------------------- WMMA GEMM ---------------------------------
// C[Md,Nd] = A * B^T in bf16, f32 accumulate.
//   !TA: A stored (Md,Kd) row-major.   TA: A stored (Kd,Md) row-major.
//   !TB: B stored (Nd,Kd) row-major.   TB: B stored (Kd,Nd) row-major.
// 128x128 block tile, BK=64, double-buffered LDS, 8 wave32 waves (2M x 4N).

enum { EPI_BF16 = 0, EPI_SILU = 1, EPI_PRED = 2, EPI_DSILU = 3, EPI_PARTIAL = 4 };

template <int EPI, bool TA, bool TB>
__global__ __launch_bounds__(256)
void gemm_wmma_bf16(const bf16_t* __restrict__ A, const bf16_t* __restrict__ Bm,
                    int Md, int Nd, int Kd, int Kc,
                    float* __restrict__ outf, bf16_t* __restrict__ outb,
                    bf16_t* __restrict__ outb2, const bf16_t* __restrict__ aux,
                    float* __restrict__ part, float gscale) {
  constexpr int BM = 128, BN = 128, BK = 64, LDT = BK + 8;
  __shared__ bf16_t sA[2][BM][LDT];
  __shared__ bf16_t sB[2][BN][LDT];

  const int tid   = threadIdx.x;
  const int lane  = tid & 31;
  const int wave  = tid >> 5;
  const int waveM = wave & 1;    // 2 waves along M (64 rows each)
  const int waveN = wave >> 1;   // 4 waves along N (32 cols each)
  const int blkM  = blockIdx.y * BM;
  const int blkN  = blockIdx.x * BN;
  const int k0    = blockIdx.z * Kc;
  const int kend  = k0 + Kc;

  v8f acc[4][2] = {};

  // ---- tile staging (BK/16 16-byte chunks per thread per matrix) ----
  auto stageA = [&](int buf, int ko) {
    if (!TA) {
#pragma unroll
      for (int c = 0; c < BK / 16; ++c) {
        int id = tid * (BK / 16) + c;
        int r = id / (BK / 8);
        int col = (id % (BK / 8)) * 8;
        const bf16_t* g = A + (size_t)(blkM + r) * Kd + ko + col;
#if HAVE_ASYNC
        async_cp16(g, &sA[buf][r][col]);
#else
        *(v8bf*)&sA[buf][r][col] = *(const v8bf*)g;
        __builtin_prefetch(g + BK, 0, 1);
#endif
      }
    } else {  // (Kd, Md) storage: scatter-transpose through VGPRs
#pragma unroll
      for (int c = 0; c < BK / 16; ++c) {
        int id = tid * (BK / 16) + c;
        int kr = id / 16;
        int col = (id % 16) * 8;
        const bf16_t* g = A + (size_t)(ko + kr) * Md + blkM + col;
        v8bf v = *(const v8bf*)g;
#pragma unroll
        for (int j = 0; j < 8; ++j) sA[buf][col + j][kr] = v[j];
        __builtin_prefetch(g + (size_t)BK * Md, 0, 1);
      }
    }
  };
  auto stageB = [&](int buf, int ko) {
    if (!TB) {
#pragma unroll
      for (int c = 0; c < BK / 16; ++c) {
        int id = tid * (BK / 16) + c;
        int r = id / (BK / 8);
        int col = (id % (BK / 8)) * 8;
        const bf16_t* g = Bm + (size_t)(blkN + r) * Kd + ko + col;
#if HAVE_ASYNC
        async_cp16(g, &sB[buf][r][col]);
#else
        *(v8bf*)&sB[buf][r][col] = *(const v8bf*)g;
        __builtin_prefetch(g + BK, 0, 1);
#endif
      }
    } else {
#pragma unroll
      for (int c = 0; c < BK / 16; ++c) {
        int id = tid * (BK / 16) + c;
        int kr = id / 16;
        int col = (id % 16) * 8;
        const bf16_t* g = Bm + (size_t)(ko + kr) * Nd + blkN + col;
        v8bf v = *(const v8bf*)g;
#pragma unroll
        for (int j = 0; j < 8; ++j) sB[buf][col + j][kr] = v[j];
        __builtin_prefetch(g + (size_t)BK * Nd, 0, 1);
      }
    }
  };

  // ---- WMMA over one staged K-tile (documented bf16 fragment layouts) ----
  auto compute = [&](int buf) {
    union FragU { v16bf v; v8bf h[2]; };
    const int rlo = lane & 15;
    const int kb  = (lane >> 4) * 8;    // A: K 0..7|16..23 / 8..15|24..31
    const int ks  = (lane >> 4) * 8;    // B: K 0..7|16..23 / 8..15|24..31
#pragma unroll
    for (int kk = 0; kk < BK; kk += 32) {
      FragU fa[4], fb[2];
#pragma unroll
      for (int mt = 0; mt < 4; ++mt) {
        int r = waveM * 64 + mt * 16 + rlo;
        fa[mt].h[0] = *(const v8bf*)&sA[buf][r][kk + kb];
        fa[mt].h[1] = *(const v8bf*)&sA[buf][r][kk + kb + 16];
      }
#pragma unroll
      for (int nt = 0; nt < 2; ++nt) {
        int cc = waveN * 32 + nt * 16 + rlo;
        fb[nt].h[0] = *(const v8bf*)&sB[buf][cc][kk + ks];
        fb[nt].h[1] = *(const v8bf*)&sB[buf][cc][kk + ks + 16];
      }
#pragma unroll
      for (int mt = 0; mt < 4; ++mt)
#pragma unroll
        for (int nt = 0; nt < 2; ++nt)
          acc[mt][nt] = __builtin_amdgcn_wmma_f32_16x16x32_bf16(
              false, fa[mt].v, false, fb[nt].v, (short)0, acc[mt][nt], false, false);
    }
  };

  // ---- double-buffered main loop: one barrier per K-step ----
  int cur = 0;
  stageA(0, k0);
  stageB(0, k0);
  for (int ko = k0; ko < kend; ko += BK) {
#if HAVE_ASYNC
    __builtin_amdgcn_s_wait_asynccnt(0);
#endif
    __syncthreads();
    if (ko + BK < kend) { stageA(cur ^ 1, ko + BK); stageB(cur ^ 1, ko + BK); }
    compute(cur);
    cur ^= 1;
  }

  // ---------------------------- fused epilogue -----------------------------
  const int lhi = lane >> 4;
#pragma unroll
  for (int mt = 0; mt < 4; ++mt) {
#pragma unroll
    for (int nt = 0; nt < 2; ++nt) {
      int gm0 = blkM + waveM * 64 + mt * 16 + lhi * 8;
      int gn  = blkN + waveN * 32 + nt * 16 + (lane & 15);
#pragma unroll
      for (int e = 0; e < 8; ++e) {
        int gm = gm0 + e;
        size_t idx = (size_t)gm * Nd + gn;
        float v = acc[mt][nt][e];
        if (EPI == EPI_BF16) {
          outb[idx] = (bf16_t)v;
        } else if (EPI == EPI_SILU) {
          float s = 1.f / (1.f + __expf(-v));
          outb[idx]  = (bf16_t)v;        // z (pre-activation, for backward)
          outb2[idx] = (bf16_t)(v * s);  // h = silu(z)
        } else if (EPI == EPI_PRED) {
          outf[idx] = v;                                  // mem_out
          float val = (float)aux[idx];                    // values (bf16)
          outb[idx] = (bf16_t)((v - val) * gscale);       // dL/dpred
        } else if (EPI == EPI_DSILU) {
          float z = (float)aux[idx];
          float s = 1.f / (1.f + __expf(-z));
          outb[idx] = (bf16_t)(v * (s * (1.f + z * (1.f - s))));
        } else {  // EPI_PARTIAL: split-K partial sums
          part[(size_t)blockIdx.z * ((size_t)Md * Nd) + idx] = v;
        }
      }
    }
  }
}

// ------------------------------- launcher ----------------------------------

extern "C" void kernel_launch(void* const* d_in, const int* in_sizes, int n_in,
                              void* d_out, int out_size, void* d_ws, size_t ws_size,
                              hipStream_t stream) {
  (void)in_sizes; (void)n_in; (void)out_size; (void)ws_size;

  const float* x     = (const float*)d_in[0];
  const float* convw = (const float*)d_in[1];
  const float* wk    = (const float*)d_in[2];
  const float* wv    = (const float*)d_in[3];
  const float* w_in  = (const float*)d_in[4];
  const float* w_h   = (const float*)d_in[5];
  const float* w_out = (const float*)d_in[6];
  const float* paw   = (const float*)d_in[7];
  const float* pab   = (const float*)d_in[8];
  const float* pew   = (const float*)d_in[9];
  const float* peb   = (const float*)d_in[10];
  const float* ptw   = (const float*)d_in[11];
  const float* ptb   = (const float*)d_in[12];
  const float* m_in  = (const float*)d_in[13];
  const float* m_h   = (const float*)d_in[14];
  const float* m_out = (const float*)d_in[15];
  float* out = (float*)d_out;

  // d_out layout (flat, reference return order)
  const size_t o_mem    = 0;
  const size_t o_nw_in  = o_mem    + (size_t)NN * DD;
  const size_t o_nw_h   = o_nw_in  + (size_t)MM * DD;
  const size_t o_nw_out = o_nw_h   + 2ull * MM * MM;
  const size_t o_nm_in  = o_nw_out + (size_t)DD * MM;
  const size_t o_nm_h   = o_nm_in  + (size_t)MM * DD;
  const size_t o_nm_out = o_nm_h   + 2ull * MM * MM;

  // ---- carve workspace (~160 MB) ----
  uint8_t* base = (uint8_t*)d_ws;
  size_t off = 0;
  auto alloc = [&](size_t bytes) -> void* {
    void* p = base + off;
    off = (off + bytes + 255) & ~(size_t)255;
    return p;
  };
  float*  scal  = (float*)alloc(4 * sizeof(float));
  float*  xpart = (float*)alloc((size_t)RCH * DD * sizeof(float));
  float*  gpart = (float*)alloc((size_t)SPLITK * DD * MM * sizeof(float));
  bf16_t* xc    = (bf16_t*)alloc((size_t)NN * DD * 2);
  bf16_t* keys  = (bf16_t*)alloc((size_t)NN * DD * 2);
  bf16_t* vals  = (bf16_t*)alloc((size_t)NN * DD * 2);
  bf16_t* dpred = (bf16_t*)alloc((size_t)NN * DD * 2);
  bf16_t* z1    = (bf16_t*)alloc((size_t)NN * MM * 2);
  bf16_t* h1    = (bf16_t*)alloc((size_t)NN * MM * 2);
  bf16_t* z2    = (bf16_t*)alloc((size_t)NN * MM * 2);
  bf16_t* h2    = (bf16_t*)alloc((size_t)NN * MM * 2);
  bf16_t* z3    = (bf16_t*)alloc((size_t)NN * MM * 2);
  bf16_t* h3    = (bf16_t*)alloc((size_t)NN * MM * 2);
  bf16_t* dz1   = (bf16_t*)alloc((size_t)NN * MM * 2);
  bf16_t* dz2   = (bf16_t*)alloc((size_t)NN * MM * 2);
  bf16_t* dz3   = (bf16_t*)alloc((size_t)NN * MM * 2);
  bf16_t* wkb   = (bf16_t*)alloc((size_t)DD * DD * 2);
  bf16_t* wvb   = (bf16_t*)alloc((size_t)DD * DD * 2);
  bf16_t* winb  = (bf16_t*)alloc((size_t)MM * DD * 2);
  bf16_t* whb   = (bf16_t*)alloc(2ull * MM * MM * 2);   // [0]=w_h0, [MM*MM]=w_h1
  bf16_t* woutb = (bf16_t*)alloc((size_t)DD * MM * 2);

  // ---- scalars + conv + weight conversion ----
  colsum_kernel<<<dim3(DD / 256, RCH), 256, 0, stream>>>(x, xpart);
  scalars_kernel<<<1, 256, 0, stream>>>(xpart, paw, pab, pew, peb, ptw, ptb, scal);
  conv_bf16_kernel<<<dim3((NN * DD) / 256), 256, 0, stream>>>(x, convw, xc);
  f32_to_bf16_kernel<<<dim3((DD * DD) / 256), 256, 0, stream>>>(wk, wkb, DD * DD);
  f32_to_bf16_kernel<<<dim3((DD * DD) / 256), 256, 0, stream>>>(wv, wvb, DD * DD);
  f32_to_bf16_kernel<<<dim3((MM * DD) / 256), 256, 0, stream>>>(w_in, winb, MM * DD);
  f32_to_bf16_kernel<<<dim3((2 * MM * MM) / 256), 256, 0, stream>>>(w_h, whb, 2 * MM * MM);
  f32_to_bf16_kernel<<<dim3((DD * MM) / 256), 256, 0, stream>>>(w_out, woutb, DD * MM);

  const dim3 blk(256);
  const float gscale = 2.f / (float)((size_t)NN * DD);
  const int Kc = NN / SPLITK;

  // ---- forward GEMMs ----
  gemm_wmma_bf16<EPI_BF16, false, false><<<dim3(DD / 128, NN / 128), blk, 0, stream>>>(
      xc, wkb, NN, DD, DD, DD, nullptr, keys, nullptr, nullptr, nullptr, 0.f);
  gemm_wmma_bf16<EPI_BF16, false, false><<<dim3(DD / 128, NN / 128), blk, 0, stream>>>(
      xc, wvb, NN, DD, DD, DD, nullptr, vals, nullptr, nullptr, nullptr, 0.f);
  gemm_wmma_bf16<EPI_SILU, false, false><<<dim3(MM / 128, NN / 128), blk, 0, stream>>>(
      keys, winb, NN, MM, DD, DD, nullptr, z1, h1, nullptr, nullptr, 0.f);
  gemm_wmma_bf16<EPI_SILU, false, false><<<dim3(MM / 128, NN / 128), blk, 0, stream>>>(
      h1, whb, NN, MM, MM, MM, nullptr, z2, h2, nullptr, nullptr, 0.f);
  gemm_wmma_bf16<EPI_SILU, false, false><<<dim3(MM / 128, NN / 128), blk, 0, stream>>>(
      h2, whb + (size_t)MM * MM, NN, MM, MM, MM, nullptr, z3, h3, nullptr, nullptr, 0.f);
  // pred = h3 @ w_out^T ; mem_out = pred ; dpred = 2*(pred - values)/(N*D)
  gemm_wmma_bf16<EPI_PRED, false, false><<<dim3(DD / 128, NN / 128), blk, 0, stream>>>(
      h3, woutb, NN, DD, MM, MM, out + o_mem, dpred, nullptr, vals, nullptr, gscale);

  // ---- backward dz chain (TRANS_B reads stored weights directly) ----
  gemm_wmma_bf16<EPI_DSILU, false, true><<<dim3(MM / 128, NN / 128), blk, 0, stream>>>(
      dpred, woutb, NN, MM, DD, DD, nullptr, dz3, nullptr, z3, nullptr, 0.f);
  gemm_wmma_bf16<EPI_DSILU, false, true><<<dim3(MM / 128, NN / 128), blk, 0, stream>>>(
      dz3, whb + (size_t)MM * MM, NN, MM, MM, MM, nullptr, dz2, nullptr, z2, nullptr, 0.f);
  gemm_wmma_bf16<EPI_DSILU, false, true><<<dim3(MM / 128, NN / 128), blk, 0, stream>>>(
      dz2, whb, NN, MM, MM, MM, nullptr, dz1, nullptr, z1, nullptr, 0.f);

  // ---- weight grads: split-K WMMA partials + fused momentum/weight update ----
  // g_out = dpred^T @ h3 -> (D, M)
  gemm_wmma_bf16<EPI_PARTIAL, true, true><<<dim3(MM / 128, DD / 128, SPLITK), blk, 0, stream>>>(
      dpred, h3, DD, MM, NN, Kc, nullptr, nullptr, nullptr, nullptr, gpart, 0.f);
  update_reduce_kernel<<<dim3((unsigned)(((size_t)DD * MM + 255) / 256)), blk, 0, stream>>>(
      gpart, (size_t)DD * MM, w_out, m_out, out + o_nw_out, out + o_nm_out, scal);
  // g_h1 = dz3^T @ h2 -> (M, M)
  gemm_wmma_bf16<EPI_PARTIAL, true, true><<<dim3(MM / 128, MM / 128, SPLITK), blk, 0, stream>>>(
      dz3, h2, MM, MM, NN, Kc, nullptr, nullptr, nullptr, nullptr, gpart, 0.f);
  update_reduce_kernel<<<dim3((unsigned)(((size_t)MM * MM + 255) / 256)), blk, 0, stream>>>(
      gpart, (size_t)MM * MM, w_h + (size_t)MM * MM, m_h + (size_t)MM * MM,
      out + o_nw_h + (size_t)MM * MM, out + o_nm_h + (size_t)MM * MM, scal);
  // g_h0 = dz2^T @ h1 -> (M, M)
  gemm_wmma_bf16<EPI_PARTIAL, true, true><<<dim3(MM / 128, MM / 128, SPLITK), blk, 0, stream>>>(
      dz2, h1, MM, MM, NN, Kc, nullptr, nullptr, nullptr, nullptr, gpart, 0.f);
  update_reduce_kernel<<<dim3((unsigned)(((size_t)MM * MM + 255) / 256)), blk, 0, stream>>>(
      gpart, (size_t)MM * MM, w_h, m_h, out + o_nw_h, out + o_nm_h, scal);
  // g_in = dz1^T @ keys -> (M, D)
  gemm_wmma_bf16<EPI_PARTIAL, true, true><<<dim3(DD / 128, MM / 128, SPLITK), blk, 0, stream>>>(
      dz1, keys, MM, DD, NN, Kc, nullptr, nullptr, nullptr, nullptr, gpart, 0.f);
  update_reduce_kernel<<<dim3((unsigned)(((size_t)MM * DD + 255) / 256)), blk, 0, stream>>>(
      gpart, (size_t)MM * DD, w_in, m_in, out + o_nw_in, out + o_nm_in, scal);
}
